// MultiHeadAttention_40467181863416
// MI455X (gfx1250) — matrix operand order repaired and checked by reading, not compiled, not run on hardware
//
#include <hip/hip_runtime.h>

// ---------------------------------------------------------------------------
// MHA forward on gfx1250 (MI455X), bf16 WMMA with f32 accumulation.
//   Pass 1 (x3): proj_kernel : P = X @ W^T + b  -> bf16, per-head layout
//   Pass 2     : attn_kernel : flash-style online-softmax attention with
//                double-buffered async K-panel staging into LDS (ASYNCcnt).
// Workspace use: 3 * B*S*D bf16 = 24 MB.
// ---------------------------------------------------------------------------

typedef __attribute__((ext_vector_type(16))) __bf16     v16bf;
typedef __attribute__((ext_vector_type(8)))  float      v8f;
typedef __attribute__((ext_vector_type(4)))  unsigned   v4u;
typedef __attribute__((ext_vector_type(4)))  int        v4i;

union FragAB { v16bf v; v4u q[2]; unsigned short h[16]; };
union FragC  { v8f   v; float f[8]; };

// Native bf16 convert (RNE) -> lets the backend use v_cvt_pk_bf16_f32.
static __device__ __forceinline__ unsigned short f2bf(float x) {
    union { __bf16 b; unsigned short u; } c;
    c.b = (__bf16)x;
    return c.u;
}

#define WMMA_BF16(A, B, C) \
    __builtin_amdgcn_wmma_f32_16x16x32_bf16(false, (A), false, (B), (short)0, (C), false, false)

#if __has_builtin(__builtin_amdgcn_global_load_async_to_lds_b128) && \
    __has_builtin(__builtin_amdgcn_s_wait_asynccnt)
#define HAVE_ASYNC_LDS 1
#else
#define HAVE_ASYNC_LDS 0
#endif

static __device__ __forceinline__ void async_wait_all() {
#if HAVE_ASYNC_LDS
    __builtin_amdgcn_s_wait_asynccnt(0);
#endif
}

// Copy one 128-byte K row (64 bf16) global -> LDS.
// Async path: 8x global_load_async_to_lds_b128 (INST_OFFSET is added to both
// the global and LDS addresses per ISA 08_async_tensor 4.4, so offset-only
// addressing walks both sides together). Builtin takes int4 pointers in
// AS1 (global) / AS3 (LDS) per the clang diagnostic.
static __device__ __forceinline__ void stage_k_row(const unsigned short* g,
                                                   unsigned short* l) {
#if HAVE_ASYNC_LDS
    typedef __attribute__((address_space(1))) v4i* gvp;
    typedef __attribute__((address_space(3))) v4i* lvp;
    gvp gp = (gvp)(void*)g;
    lvp lp = (lvp)(void*)l;
    __builtin_amdgcn_global_load_async_to_lds_b128(gp, lp,   0, 0);
    __builtin_amdgcn_global_load_async_to_lds_b128(gp, lp,  16, 0);
    __builtin_amdgcn_global_load_async_to_lds_b128(gp, lp,  32, 0);
    __builtin_amdgcn_global_load_async_to_lds_b128(gp, lp,  48, 0);
    __builtin_amdgcn_global_load_async_to_lds_b128(gp, lp,  64, 0);
    __builtin_amdgcn_global_load_async_to_lds_b128(gp, lp,  80, 0);
    __builtin_amdgcn_global_load_async_to_lds_b128(gp, lp,  96, 0);
    __builtin_amdgcn_global_load_async_to_lds_b128(gp, lp, 112, 0);
#else
    #pragma unroll
    for (int j = 0; j < 8; ++j)
        ((v4u*)l)[j] = ((const v4u*)g)[j];
#endif
}

constexpr int Bn = 2, S = 2048, Dm = 1024, H = 16, HD = 64;

// ---------------------------------------------------------------------------
// Projection GEMM: out = X(4096x1024) @ W^T(1024x1024) + bias, bf16 output.
// One wave -> 16(M) x 64(N) strip, A fragment reused across 4 N-subtiles.
// TRANS=0: out[b][h][s][d]   (for Q, K)
// TRANS=1: out[b][h][d][s]   (for V, so PV B-fragments are key-contiguous)
// ---------------------------------------------------------------------------
template<int TRANS>
__global__ __launch_bounds__(256) void proj_kernel(
    const float* __restrict__ X, const float* __restrict__ W,
    const float* __restrict__ bias, unsigned short* __restrict__ out)
{
    const int tid  = blockIdx.x * blockDim.x + threadIdx.x;
    const int wid  = tid >> 5;
    const int lane = tid & 31;
    const int half = lane >> 4, lr = lane & 15;
    const int m0   = (wid >> 4) * 16;     // 256 row tiles
    const int n0g  = (wid & 15) * 64;     // 16 col groups of 64

    FragC acc[4];
    #pragma unroll
    for (int t = 0; t < 4; ++t)
        #pragma unroll
        for (int r = 0; r < 8; ++r) acc[t].f[r] = 0.f;

    const float* arow = X + (size_t)(m0 + lr) * Dm;
    const int k0 = 16 * half, k1 = 16 * half + 8;   // A-layout K split per lane half

    for (int kk = 0; kk < Dm; kk += 32) {
        FragAB a;
        #pragma unroll
        for (int j = 0; j < 8; ++j) {
            a.h[j]     = f2bf(arow[kk + k0 + j]);
            a.h[8 + j] = f2bf(arow[kk + k1 + j]);
        }
        #pragma unroll
        for (int t = 0; t < 4; ++t) {
            FragAB bf;
            const float* wrow = W + (size_t)(n0g + t * 16 + lr) * Dm + kk + 16 * half;
            #pragma unroll
            for (int j = 0; j < 16; ++j) bf.h[j] = f2bf(wrow[j]);
            acc[t].v = WMMA_BF16(a.v, bf.v, acc[t].v);
        }
    }

    #pragma unroll
    for (int t = 0; t < 4; ++t) {
        const int n  = n0g + t * 16 + lr;
        const float bv = bias[n];
        const int hh = n >> 6, d = n & (HD - 1);
        #pragma unroll
        for (int r = 0; r < 8; ++r) {
            const int m  = m0 + r + 8 * half;
            const int bb = m >> 11, s = m & (S - 1);
            size_t idx;
            if (TRANS) idx = (((size_t)bb * H + hh) * HD + d) * S + s;
            else       idx = (((size_t)bb * H + hh) * S + s) * HD + d;
            out[idx] = f2bf(acc[t].f[r] + bv);
        }
    }
}

// ---------------------------------------------------------------------------
// Attention: one wave per (b, h, 16 query rows); online softmax over S keys
// in steps of 32. K panels double-buffered in LDS via async copies; per step:
// 4 QK WMMAs + LDS transpose of P + 4 PV WMMAs.
// ---------------------------------------------------------------------------
__global__ __launch_bounds__(128) void attn_kernel(
    const unsigned short* __restrict__ qh,   // [B,H,S,64] bf16
    const unsigned short* __restrict__ kh,   // [B,H,S,64] bf16
    const unsigned short* __restrict__ vtr,  // [B,H,64,S] bf16
    const float* __restrict__ mask,          // [S,S] f32
    float* __restrict__ out)                 // [B,S,D] f32
{
    __shared__ unsigned short kbuf[4 * 2 * 32 * HD];  // 32 KB: per-wave 2x K panels
    __shared__ unsigned short pbuf[4 * 16 * 32];      //  4 KB: per-wave P slab

    const int tid  = blockIdx.x * blockDim.x + threadIdx.x;
    const int wid  = tid >> 5;
    const int lane = tid & 31;
    const int half = lane >> 4, lr = lane & 15;
    const int wib  = threadIdx.x >> 5;

    const int qt = wid & 127;         // S/16 query tiles
    const int bh = wid >> 7;          // 32 (b,h) pairs
    const int b  = bh >> 4, h = bh & 15;
    const int q0 = qt * 16;

    unsigned short* kslab = kbuf + wib * (2 * 32 * HD);
    unsigned short* myl   = pbuf + wib * 512;
    const unsigned short* kpanel = kh + (size_t)bh * S * HD;

    // Q fragments (A-layout): rows q0..q0+15, K = d in [0,32) and [32,64)
    const unsigned short* qrow = qh + ((size_t)bh * S + q0 + lr) * HD;
    FragAB a0, a1;
    a0.q[0] = *(const v4u*)(qrow +      16 * half);
    a0.q[1] = *(const v4u*)(qrow +  8 + 16 * half);
    a1.q[0] = *(const v4u*)(qrow + 32 + 16 * half);
    a1.q[1] = *(const v4u*)(qrow + 40 + 16 * half);

    FragC o[4];
    float rmax[8], rsum[8];
    #pragma unroll
    for (int r = 0; r < 8; ++r) { rmax[r] = -3.0e38f; rsum[r] = 0.f; }
    #pragma unroll
    for (int dc = 0; dc < 4; ++dc)
        #pragma unroll
        for (int r = 0; r < 8; ++r) o[dc].f[r] = 0.f;

    // Preload K panel 0 (each lane stages one 128-byte K row).
    stage_k_row(kpanel + (size_t)lane * HD, kslab + lane * HD);

    int cur = 0;
    for (int kt = 0; kt < S; kt += 32) {
        async_wait_all();                         // s_wait_asynccnt 0
        const unsigned short* kc = kslab + cur * (32 * HD);

        // Kick off async staging of the next K panel into the other slab;
        // it overlaps all WMMA/softmax work of this iteration.
        if (kt + 32 < S)
            stage_k_row(kpanel + (size_t)(kt + 32 + lane) * HD,
                        kslab + (cur ^ 1) * (32 * HD) + lane * HD);

        // ---- scores: two 16x16 tiles over keys [kt,kt+16) and [kt+16,kt+32)
        const unsigned short* kr0 = kc + lr * HD + 16 * half;          // keys kt+lr
        const unsigned short* kr1 = kc + (16 + lr) * HD + 16 * half;   // keys kt+16+lr

        FragC s0, s1;
        #pragma unroll
        for (int r = 0; r < 8; ++r) { s0.f[r] = 0.f; s1.f[r] = 0.f; }
        {
            FragAB bk;
            bk.q[0] = *(const v4u*)(kr0);      bk.q[1] = *(const v4u*)(kr0 + 8);
            s0.v = WMMA_BF16(a0.v, bk.v, s0.v);
            bk.q[0] = *(const v4u*)(kr0 + 32); bk.q[1] = *(const v4u*)(kr0 + 40);
            s0.v = WMMA_BF16(a1.v, bk.v, s0.v);
            bk.q[0] = *(const v4u*)(kr1);      bk.q[1] = *(const v4u*)(kr1 + 8);
            s1.v = WMMA_BF16(a0.v, bk.v, s1.v);
            bk.q[0] = *(const v4u*)(kr1 + 32); bk.q[1] = *(const v4u*)(kr1 + 40);
            s1.v = WMMA_BF16(a1.v, bk.v, s1.v);
        }

        // ---- scale 1/sqrt(64), additive mask, online softmax update
        #pragma unroll
        for (int r = 0; r < 8; ++r) {
            const size_t mro = (size_t)(q0 + r + 8 * half) * S + kt;
            s0.f[r] = s0.f[r] * 0.125f + mask[mro + lr];
            s1.f[r] = s1.f[r] * 0.125f + mask[mro + 16 + lr];

            float lm = fmaxf(s0.f[r], s1.f[r]);
            lm = fmaxf(lm, __shfl_xor(lm, 1));
            lm = fmaxf(lm, __shfl_xor(lm, 2));
            lm = fmaxf(lm, __shfl_xor(lm, 4));
            lm = fmaxf(lm, __shfl_xor(lm, 8));   // row max within 16-lane half

            const float nm = fmaxf(rmax[r], lm);
            const float sc = __expf(rmax[r] - nm);
            rmax[r] = nm;

            const float p0 = __expf(s0.f[r] - nm);
            const float p1 = __expf(s1.f[r] - nm);
            s0.f[r] = p0; s1.f[r] = p1;

            float ps = p0 + p1;
            ps += __shfl_xor(ps, 1);
            ps += __shfl_xor(ps, 2);
            ps += __shfl_xor(ps, 4);
            ps += __shfl_xor(ps, 8);             // row sum within 16-lane half
            rsum[r] = rsum[r] * sc + ps;

            #pragma unroll
            for (int dc = 0; dc < 4; ++dc) o[dc].f[r] *= sc;
        }

        // ---- C-layout -> A-layout transpose of P through LDS (bf16).
        // DS ops from one wave execute in order (ISA 7.1), no barrier needed:
        // the wave's own stores complete before its loads read them.
        #pragma unroll
        for (int r = 0; r < 8; ++r) {
            myl[(r + 8 * half) * 32 + lr]      = f2bf(s0.f[r]);
            myl[(r + 8 * half) * 32 + 16 + lr] = f2bf(s1.f[r]);
        }
        FragAB pa;
    pa.q[0] = *(const v4u*)(myl + lr * 32 +     16 * half);
    pa.q[1] = *(const v4u*)(myl + lr * 32 + 8 + 16 * half);

        // ---- PV: 4 d-chunks of 16; V^T rows are key-contiguous
        #pragma unroll
        for (int dc = 0; dc < 4; ++dc) {
            FragAB bv;
            const unsigned short* vrow =
                vtr + ((size_t)bh * HD + dc * 16 + lr) * S + kt + 16 * half;
            if (kt + 32 < S)
                __builtin_prefetch(vrow + 32, 0, 1);   // global_prefetch_b8
            bv.q[0] = *(const v4u*)(vrow);
            bv.q[1] = *(const v4u*)(vrow + 8);
            o[dc].v = WMMA_BF16(pa.v, bv.v, o[dc].v);
        }

        cur ^= 1;
    }

    // ---- normalize and store fp32 (B, S, D)
    #pragma unroll
    for (int dc = 0; dc < 4; ++dc)
        #pragma unroll
        for (int r = 0; r < 8; ++r) {
            const int m = q0 + r + 8 * half;
            out[((size_t)b * S + m) * Dm + h * HD + dc * 16 + lr] =
                o[dc].f[r] / rsum[r];
        }
}

// ---------------------------------------------------------------------------
extern "C" void kernel_launch(void* const* d_in, const int* in_sizes, int n_in,
                              void* d_out, int out_size, void* d_ws, size_t ws_size,
                              hipStream_t stream)
{
    const float* q    = (const float*)d_in[0];
    const float* k    = (const float*)d_in[1];
    const float* v    = (const float*)d_in[2];
    const float* mask = (const float*)d_in[3];
    const float* Wq   = (const float*)d_in[4];
    const float* bq   = (const float*)d_in[5];
    const float* Wk   = (const float*)d_in[6];
    const float* bk   = (const float*)d_in[7];
    const float* Wv   = (const float*)d_in[8];
    const float* bv   = (const float*)d_in[9];

    const size_t nE = (size_t)Bn * S * Dm;           // 4M bf16 elems per tensor
    unsigned short* qh  = (unsigned short*)d_ws;     // 8 MB
    unsigned short* kh  = qh + nE;                   // 8 MB
    unsigned short* vtr = kh + nE;                   // 8 MB (transposed V)

    // 4096 waves each (256 M-tiles x 16 N-groups); 8 waves / 256-thread block
    proj_kernel<0><<<512, 256, 0, stream>>>(q, Wq, bq, qh);
    proj_kernel<0><<<512, 256, 0, stream>>>(k, Wk, bk, kh);
    proj_kernel<1><<<512, 256, 0, stream>>>(v, Wv, bv, vtr);

    // 2*16*128 = 4096 waves; 4 waves / 128-thread block (36 KB LDS/block)
    attn_kernel<<<1024, 128, 0, stream>>>(qh, kh, vtr, mask, (float*)d_out);
}
